// VisualAdapter_21603685499007
// MI455X (gfx1250) — matrix operand order repaired and checked by reading, not compiled, not run on hardware
//
#include <hip/hip_runtime.h>
#include <math.h>

typedef __attribute__((ext_vector_type(16))) __bf16 v16bf;
typedef __attribute__((ext_vector_type(2)))  __bf16 v2bf;
typedef __attribute__((ext_vector_type(8)))  float  v8f;
typedef __attribute__((ext_vector_type(2)))  float  vf2;

namespace {
constexpr int Bn  = 160;
constexpr int Ln  = 196;
constexpr int Cn  = 768;
constexpr int Dn  = 512;
constexpr int NVn = 196;
constexpr int Tn  = 87;
constexpr int DMn = Cn / 2;    // 384
constexpr int DSn = Cn / 16;   // 48
constexpr int NTQ = DMn / 16;  // 24 n-tiles worth of partials in the cq GEMM
constexpr float EPSf   = 1e-5f;
constexpr float ALPHAf = 0.3f;
constexpr float BETAf  = 0.05f;
constexpr float GAMMAf = 0.05f;
}

__device__ __forceinline__ v2bf cvt2(float f0, float f1) {
  vf2 f; f[0] = f0; f[1] = f1;
  return __builtin_convertvector(f, v2bf);
}

// ---------------- WMMA fragment loaders ----------------------------------
// Two-phase: (1) issue all 16 independent loads into a raw array, (2) convert.
// This lets the scheduler batch the loads under one wait instead of
// load->wait->cvt per pair.  Guards are template flags so interior tiles
// compile with zero selects.
// A fragment: 16x32 bf16, element A[m][k] at base + m*sm + k*sk
template <bool RG, bool KG>
__device__ __forceinline__ v16bf load_frag_a(const float* __restrict__ A, int sm, int sk,
                                             int M, int K, int m0, int kb) {
  int lane = threadIdx.x & 31;
  int h = lane >> 4, mm = lane & 15;
  int row = m0 + mm;
  bool rok = !RG || (row < M);
  int rowc = RG ? (rok ? row : (M - 1)) : row;
  const float* __restrict__ ap = A + (size_t)rowc * sm;
  float f[16];
#pragma unroll
  for (int j = 0; j < 8; ++j) {
    int k0 = kb + ((j < 4) ? (2 * j) : (16 + 2 * (j - 4))) + 8 * h;
    int k1 = k0 + 1;
    int k0c = k0, k1c = k1;
    if (KG) { k0c = (k0 < K) ? k0 : (K - 1); k1c = (k1 < K) ? k1 : (K - 1); }
    f[2 * j]     = ap[(size_t)k0c * sk];   // unconditional loads (clamped addr)
    f[2 * j + 1] = ap[(size_t)k1c * sk];
  }
  if (RG || KG) {
#pragma unroll
    for (int j = 0; j < 8; ++j) {
      int k0 = kb + ((j < 4) ? (2 * j) : (16 + 2 * (j - 4))) + 8 * h;
      bool ok0 = rok && (!KG || (k0 < K));
      bool ok1 = rok && (!KG || (k0 + 1 < K));
      f[2 * j]     = ok0 ? f[2 * j]     : 0.0f;
      f[2 * j + 1] = ok1 ? f[2 * j + 1] : 0.0f;
    }
  }
  v16bf a;
#pragma unroll
  for (int j = 0; j < 8; ++j) {
    v2bf p = cvt2(f[2 * j], f[2 * j + 1]);
    a[2 * j]     = p[0];
    a[2 * j + 1] = p[1];
  }
  return a;
}

// B fragment: 32x16 bf16, element B[k][n] at base + k*sk + n*sn
template <bool CG, bool KG>
__device__ __forceinline__ v16bf load_frag_b(const float* __restrict__ Bp, int sk, int sn,
                                             int K, int N, int kb, int n0) {
  int lane = threadIdx.x & 31;
  int h = lane >> 4, nn = lane & 15;
  int col = n0 + nn;
  bool cok = !CG || (col < N);
  int colc = CG ? (cok ? col : (N - 1)) : col;
  const float* __restrict__ bp = Bp + (size_t)colc * sn;
  float f[16];
#pragma unroll
  for (int j = 0; j < 8; ++j) {
    int k0 = kb + ((j < 4) ? (2 * j) : (16 + 2 * (j - 4))) + 8 * h;
    int k1 = k0 + 1;
    int k0c = k0, k1c = k1;
    if (KG) { k0c = (k0 < K) ? k0 : (K - 1); k1c = (k1 < K) ? k1 : (K - 1); }
    f[2 * j]     = bp[(size_t)k0c * sk];
    f[2 * j + 1] = bp[(size_t)k1c * sk];
  }
  if (CG || KG) {
#pragma unroll
    for (int j = 0; j < 8; ++j) {
      int k0 = kb + ((j < 4) ? (2 * j) : (16 + 2 * (j - 4))) + 8 * h;
      bool ok0 = cok && (!KG || (k0 < K));
      bool ok1 = cok && (!KG || (k0 + 1 < K));
      f[2 * j]     = ok0 ? f[2 * j]     : 0.0f;
      f[2 * j + 1] = ok1 ? f[2 * j + 1] : 0.0f;
    }
  }
  v16bf b;
#pragma unroll
  for (int j = 0; j < 8; ++j) {
    v2bf p = cvt2(f[2 * j], f[2 * j + 1]);
    b[2 * j]     = p[0];
    b[2 * j + 1] = p[1];
  }
  return b;
}

// One wave computes a (16*MB) x (16*NB) block: MB A-fragments are reused
// against each B-fragment (MB*NB WMMAs per k-step).  K is a literal at every
// call site so the tail iteration constant-folds away when K % 32 == 0.
template <int MB, int NB, bool RG, bool CG>
__device__ __forceinline__ void gemm_multi(const float* __restrict__ A, int asm_, int ask, int M,
                                           const float* __restrict__ Bp, int bsk, int bsn, int N,
                                           int K, int m0, int n0, v8f* acc) {
  int KF = K & ~31;
  for (int kb = 0; kb < KF; kb += 32) {
    v16bf a[MB];
#pragma unroll
    for (int mi = 0; mi < MB; ++mi)
      a[mi] = load_frag_a<RG, false>(A, asm_, ask, M, K, m0 + 16 * mi, kb);
#pragma unroll
    for (int nb = 0; nb < NB; ++nb) {
      v16bf b = load_frag_b<CG, false>(Bp, bsk, bsn, K, N, kb, n0 + 16 * nb);
#pragma unroll
      for (int mi = 0; mi < MB; ++mi)
        acc[mi * NB + nb] = __builtin_amdgcn_wmma_f32_16x16x32_bf16(
            false, a[mi], false, b, (short)0, acc[mi * NB + nb], false, false);
    }
  }
  if (K & 31) {
    v16bf a[MB];
#pragma unroll
    for (int mi = 0; mi < MB; ++mi)
      a[mi] = load_frag_a<RG, true>(A, asm_, ask, M, K, m0 + 16 * mi, KF);
#pragma unroll
    for (int nb = 0; nb < NB; ++nb) {
      v16bf b = load_frag_b<CG, true>(Bp, bsk, bsn, K, N, KF, n0 + 16 * nb);
#pragma unroll
      for (int mi = 0; mi < MB; ++mi)
        acc[mi * NB + nb] = __builtin_amdgcn_wmma_f32_16x16x32_bf16(
            false, a[mi], false, b, (short)0, acc[mi * NB + nb], false, false);
    }
  }
}

// Wave-uniform dispatch: interior blocks take the select-free fast path.
template <int MB, int NB>
__device__ __forceinline__ void gemm_dispatch(const float* __restrict__ A, int asm_, int ask, int M,
                                              const float* __restrict__ Bp, int bsk, int bsn, int N,
                                              int K, int m0, int n0, v8f* acc) {
  bool rg = (m0 + 16 * MB > M);
  bool cg = (n0 + 16 * NB > N);
  if (!rg && !cg)
    gemm_multi<MB, NB, false, false>(A, asm_, ask, M, Bp, bsk, bsn, N, K, m0, n0, acc);
  else
    gemm_multi<MB, NB, true, true>(A, asm_, ask, M, Bp, bsk, bsn, N, K, m0, n0, acc);
}

template <int MB, int NB, class F>
__device__ __forceinline__ void store_tiles(const v8f* acc, int m0, int n0,
                                            int M, int N, F&& f) {
  int lane = threadIdx.x & 31, h = lane >> 4, nn = lane & 15;
#pragma unroll
  for (int mi = 0; mi < MB; ++mi) {
#pragma unroll
    for (int nb = 0; nb < NB; ++nb) {
      int col = n0 + 16 * nb + nn;
      if (col < N) {
#pragma unroll
        for (int j = 0; j < 8; ++j) {
          int row = m0 + 16 * mi + j + 8 * h;
          if (row < M) f(row, col, acc[mi * NB + nb][j]);
        }
      }
    }
  }
}

// ================= GEMM kernels (blockDim = 32, one wave per block) =======

// H[b,o,n] = sum_d conv_w[o,d] * vt[b,n,d] + conv_b[o]
__global__ void k_gemm_h(const float* __restrict__ conv_w, const float* __restrict__ conv_b,
                         const float* __restrict__ vt, float* __restrict__ H) {
  int b = blockIdx.z, m0 = blockIdx.x * 32, n0 = blockIdx.y * 64;
  const float* Bb = vt + (size_t)b * NVn * Dn;
  v8f acc[8] = {};
  gemm_dispatch<2, 4>(conv_w, Dn, 1, NVn, Bb, 1, Dn, NVn, Dn, m0, n0, acc);
  float* Cp = H + (size_t)b * NVn * NVn;
  store_tiles<2, 4>(acc, m0, n0, NVn, NVn, [&](int r, int c, float v) {
    Cp[(size_t)r * NVn + c] = v + conv_b[r];
  });
}

// V[b,o,c] = sum_n H[b,o,n] * fc_w[c,n] + fc_b[c]
__global__ void k_gemm_V(const float* __restrict__ H, const float* __restrict__ fc_w,
                         const float* __restrict__ fc_b, float* __restrict__ V) {
  int b = blockIdx.z, m0 = blockIdx.x * 32, n0 = blockIdx.y * 64;
  const float* Ab = H + (size_t)b * NVn * NVn;
  v8f acc[8] = {};
  gemm_dispatch<2, 4>(Ab, NVn, 1, NVn, fc_w, 1, NVn, Cn, NVn, m0, n0, acc);
  float* Cp = V + (size_t)b * NVn * Cn;
  store_tiles<2, 4>(acc, m0, n0, NVn, Cn, [&](int r, int c, float v) {
    Cp[(size_t)r * Cn + c] = v + fc_b[c];
  });
}

// scores[b,t,n] = sum_c my_tokens[t,c] * V[b,n,c]
__global__ void k_gemm_s1(const float* __restrict__ mtk, const float* __restrict__ V,
                          float* __restrict__ S) {
  int b = blockIdx.z, m0 = blockIdx.x * 32, n0 = blockIdx.y * 64;
  const float* Bb = V + (size_t)b * NVn * Cn;
  v8f acc[8] = {};
  gemm_dispatch<2, 4>(mtk, Cn, 1, Tn, Bb, 1, Cn, NVn, Cn, m0, n0, acc);
  float* Cp = S + (size_t)b * Tn * NVn;
  store_tiles<2, 4>(acc, m0, n0, Tn, NVn, [&](int r, int c, float v) {
    Cp[(size_t)r * NVn + c] = v;
  });
}

// rep[b,t,c] = my_tokens[t,c] + sum_n probs[b,t,n] * V[b,n,c]
__global__ void k_gemm_rep(const float* __restrict__ P, const float* __restrict__ V,
                           const float* __restrict__ mtk, float* __restrict__ rep) {
  int b = blockIdx.z, m0 = blockIdx.x * 32, n0 = blockIdx.y * 64;
  const float* Ab = P + (size_t)b * Tn * NVn;
  const float* Bb = V + (size_t)b * NVn * Cn;
  v8f acc[8] = {};
  gemm_dispatch<2, 4>(Ab, NVn, 1, Tn, Bb, Cn, 1, Cn, NVn, m0, n0, acc);
  float* Cp = rep + (size_t)b * Tn * Cn;
  store_tiles<2, 4>(acc, m0, n0, Tn, Cn, [&](int r, int c, float v) {
    Cp[(size_t)r * Cn + c] = v + mtk[(size_t)r * Cn + c];
  });
}

// scores2[b,l,t] = sum_c x[b,c,l] * rep[b,t,c]
__global__ void k_gemm_s2(const float* __restrict__ x, const float* __restrict__ rep,
                          float* __restrict__ S) {
  int b = blockIdx.z, m0 = blockIdx.x * 32, n0 = blockIdx.y * 32;
  const float* Ab = x + (size_t)b * Cn * Ln;     // A[m=l][k=c]: sm=1, sk=Ln
  const float* Bb = rep + (size_t)b * Tn * Cn;   // B[k=c][n=t]: sk=1, sn=Cn
  v8f acc[4] = {};
  gemm_dispatch<2, 2>(Ab, 1, Ln, Ln, Bb, 1, Cn, Tn, Cn, m0, n0, acc);
  float* Cp = S + (size_t)b * Ln * Tn;
  store_tiles<2, 2>(acc, m0, n0, Ln, Tn, [&](int r, int c, float v) {
    Cp[(size_t)r * Tn + c] = v;
  });
}

// xt[b,l,c] = x[b,c,l] + gate_av * sum_t probs2[b,l,t] * rep[b,t,c]
__global__ void k_gemm_xt(const float* __restrict__ P, const float* __restrict__ rep,
                          const float* __restrict__ x, const float* __restrict__ gate_av,
                          float* __restrict__ xt) {
  int b = blockIdx.z, m0 = blockIdx.x * 32, n0 = blockIdx.y * 64;
  const float* Ab = P + (size_t)b * Ln * Tn;
  const float* Bb = rep + (size_t)b * Tn * Cn;
  v8f acc[8] = {};
  gemm_dispatch<2, 4>(Ab, Tn, 1, Ln, Bb, Cn, 1, Cn, Tn, m0, n0, acc);
  float gav = gate_av[0];
  float* Cp = xt + (size_t)b * Ln * Cn;
  const float* xb = x + (size_t)b * Cn * Ln;
  store_tiles<2, 4>(acc, m0, n0, Ln, Cn, [&](int r, int c, float v) {
    Cp[(size_t)r * Cn + c] = xb[(size_t)c * Ln + r] + gav * v;
  });
}

// vq1[b,l,c] = relu(sum_ci xt[b,l,ci] * v1_w[c,ci] + v1_b[c])
__global__ void k_gemm_vq1(const float* __restrict__ xt, const float* __restrict__ v1_w,
                           const float* __restrict__ v1_b, float* __restrict__ vq1) {
  int b = blockIdx.z, m0 = blockIdx.x * 32, n0 = blockIdx.y * 64;
  const float* Ab = xt + (size_t)b * Ln * Cn;
  v8f acc[8] = {};
  gemm_dispatch<2, 4>(Ab, Cn, 1, Ln, v1_w, 1, Cn, Cn, Cn, m0, n0, acc);
  float* Cp = vq1 + (size_t)b * Ln * Cn;
  store_tiles<2, 4>(acc, m0, n0, Ln, Cn, [&](int r, int c, float v) {
    Cp[(size_t)r * Cn + c] = fmaxf(v + v1_b[c], 0.0f);
  });
}

// cq = relu(cfeat @ v2^T + v2_b); part[b,l,nt] = sum_{dm in 16-col tile} cq*aq2*vs_w
__global__ void k_gemm_cq(const float* __restrict__ cfeat, const float* __restrict__ v2_w,
                          const float* __restrict__ v2_b, const float* __restrict__ aq2,
                          const float* __restrict__ vs_w, float* __restrict__ part) {
  int b = blockIdx.z, m0 = blockIdx.x * 32, n0 = blockIdx.y * 64;
  const float* Ab = cfeat + (size_t)b * Ln * Cn;
  v8f acc[8] = {};
  gemm_dispatch<2, 4>(Ab, Cn, 1, Ln, v2_w, 1, Cn, DMn, Cn, m0, n0, acc);
  int lane = threadIdx.x & 31, h = lane >> 4, nn = lane & 15;
#pragma unroll
  for (int mi = 0; mi < 2; ++mi) {
#pragma unroll
    for (int nb = 0; nb < 4; ++nb) {
      int col = n0 + 16 * nb + nn;               // always < DMn (384 = 6*64)
      float w = aq2[(size_t)b * DMn + col] * vs_w[col];
      float s[8];
#pragma unroll
      for (int j = 0; j < 8; ++j)
        s[j] = fmaxf(acc[mi * 4 + nb][j] + v2_b[col], 0.0f) * w;
      // reduce across the 16 lanes of each half (same rows, distinct dm cols)
#pragma unroll
      for (int j = 0; j < 8; ++j)
        for (int m = 1; m < 16; m <<= 1) s[j] += __shfl_xor(s[j], m, 32);
      if ((lane & 15) == 0) {
#pragma unroll
        for (int j = 0; j < 8; ++j) {
          int row = m0 + 16 * mi + j + 8 * h;
          if (row < Ln)
            part[((size_t)b * Ln + row) * NTQ + (blockIdx.y * 4 + nb)] = s[j];
        }
      }
    }
  }
}

// z[b,l,s] = relu(BN1(sum_c x6ln[b,l,c] * down_w[s,c]))
__global__ void k_gemm_down(const float* __restrict__ x6ln, const float* __restrict__ down_w,
                            const float* __restrict__ g, const float* __restrict__ bb,
                            const float* __restrict__ mn, const float* __restrict__ vr,
                            float* __restrict__ z) {
  int b = blockIdx.z, m0 = blockIdx.x * 32, n0 = 0;
  const float* Ab = x6ln + (size_t)b * Ln * Cn;
  v8f acc[6] = {};
  gemm_dispatch<2, 3>(Ab, Cn, 1, Ln, down_w, 1, Cn, DSn, Cn, m0, n0, acc);
  float* Cp = z + (size_t)b * Ln * DSn;
  store_tiles<2, 3>(acc, m0, n0, Ln, DSn, [&](int r, int c, float v) {
    float sc = rsqrtf(vr[c] + EPSf) * g[c];
    Cp[(size_t)r * DSn + c] = fmaxf((v - mn[c]) * sc + bb[c], 0.0f);
  });
}

// o[b,l,c] = BN2(sum_s z[b,l,s] * up_w[c,s])
__global__ void k_gemm_up(const float* __restrict__ z, const float* __restrict__ up_w,
                          const float* __restrict__ g, const float* __restrict__ bb,
                          const float* __restrict__ mn, const float* __restrict__ vr,
                          float* __restrict__ o) {
  int b = blockIdx.z, m0 = blockIdx.x * 32, n0 = blockIdx.y * 64;
  const float* Ab = z + (size_t)b * Ln * DSn;
  v8f acc[8] = {};
  gemm_dispatch<2, 4>(Ab, DSn, 1, Ln, up_w, 1, DSn, Cn, DSn, m0, n0, acc);
  float* Cp = o + (size_t)b * Ln * Cn;
  store_tiles<2, 4>(acc, m0, n0, Ln, Cn, [&](int r, int c, float v) {
    float sc = rsqrtf(vr[c] + EPSf) * g[c];
    Cp[(size_t)r * Cn + c] = (v - mn[c]) * sc + bb[c];
  });
}

// ================= elementwise / reduction kernels ========================

__global__ void k_colmean(const float* in, float* out, int R, int Ccols, float scale) {
  int idx = blockIdx.x * blockDim.x + threadIdx.x;
  if (idx >= Bn * Ccols) return;
  int b = idx / Ccols, c = idx % Ccols;
  const float* p = in + (size_t)b * R * Ccols + c;
  float s = 0.0f;
  for (int r = 0; r < R; ++r) s += p[(size_t)r * Ccols];
  out[idx] = s * scale;
}

// out[b,o] = act(sum_i in[b,i]*w[o,i] + bias[o]); act: 0=none 1=relu 2=sigmoid
__global__ void k_dense(const float* in, const float* w, const float* bias,
                        float* out, int I, int O, int act) {
  int idx = blockIdx.x * blockDim.x + threadIdx.x;
  if (idx >= Bn * O) return;
  int b = idx / O, o = idx % O;
  const float* ip = in + (size_t)b * I;
  const float* wp = w + (size_t)o * I;
  float s = bias[o];
  for (int i = 0; i < I; ++i) s += ip[i] * wp[i];
  if (act == 1) s = fmaxf(s, 0.0f);
  else if (act == 2) s = 1.0f / (1.0f + expf(-s));
  out[idx] = s;
}

__global__ void k_prod(const float* a, const float* b, float* o, int n) {
  int i = blockIdx.x * blockDim.x + threadIdx.x;
  if (i < n) o[i] = a[i] * b[i];
}

__global__ void k_copy(const float* s, float* d, int n) {
  int i = blockIdx.x * blockDim.x + threadIdx.x;
  if (i < n) d[i] = s[i];
}

// in-place row softmax, ncols <= 256
__global__ void k_softmax(float* data, int ncols) {
  __shared__ float sh[256];
  int row = blockIdx.x, t = threadIdx.x;
  float* p = data + (size_t)row * ncols;
  float v = (t < ncols) ? p[t] : -3.4e38f;
  sh[t] = v; __syncthreads();
  for (int s = 128; s > 0; s >>= 1) { if (t < s) sh[t] = fmaxf(sh[t], sh[t + s]); __syncthreads(); }
  float m = sh[0]; __syncthreads();
  float e = (t < ncols) ? expf(v - m) : 0.0f;
  sh[t] = e; __syncthreads();
  for (int s = 128; s > 0; s >>= 1) { if (t < s) sh[t] += sh[t + s]; __syncthreads(); }
  float sum = sh[0];
  if (t < ncols) p[t] = e / sum;
}

// cfeat[b,l,c] = xt[b,l,c] * (ch[b,c] + 1)
__global__ void k_cfeat(const float* xt, const float* ch, float* cf) {
  int i = blockIdx.x * blockDim.x + threadIdx.x;
  if (i >= Bn * Ln * Cn) return;
  int c = i % Cn;
  int b = i / (Ln * Cn);
  cf[i] = xt[i] * (ch[(size_t)b * Cn + c] + 1.0f);
}

// s_tmp[b,l] = vs_b + sum over NTQ partials
__global__ void k_stmp(const float* part, const float* vs_b, float* s_tmp) {
  int i = blockIdx.x * blockDim.x + threadIdx.x;
  if (i >= Bn * Ln) return;
  float s = vs_b[0];
  const float* p = part + (size_t)i * NTQ;
#pragma unroll
  for (int j = 0; j < NTQ; ++j) s += p[j];
  s_tmp[i] = s;
}

// per-b: s_sig = sigmoid(s_tmp); spatial = softmax_L(tanh(s_tmp)) -> out
__global__ void k_spatial(const float* s_tmp, float* s_sig, float* out_sp) {
  __shared__ float sh[256];
  int b = blockIdx.x, t = threadIdx.x;
  float v = (t < Ln) ? s_tmp[(size_t)b * Ln + t] : 0.0f;
  if (t < Ln) s_sig[(size_t)b * Ln + t] = 1.0f / (1.0f + expf(-v));
  float th = (t < Ln) ? tanhf(v) : -3.4e38f;
  sh[t] = th; __syncthreads();
  for (int s = 128; s > 0; s >>= 1) { if (t < s) sh[t] = fmaxf(sh[t], sh[t + s]); __syncthreads(); }
  float m = sh[0]; __syncthreads();
  float e = (t < Ln) ? expf(th - m) : 0.0f;
  sh[t] = e; __syncthreads();
  for (int s = 128; s > 0; s >>= 1) { if (t < s) sh[t] += sh[t + s]; __syncthreads(); }
  float sum = sh[0];
  if (t < Ln) out_sp[(size_t)b * Ln + t] = e / sum;
}

// x6 = xt * factor; x6ln = LayerNorm_C(x6) * g + b   (one block per (b,l) row)
__global__ void k_x6ln(const float* xt, const float* ch, const float* s_sig,
                       const float* tmprl, const float* g, const float* bb,
                       float* out) {
  __shared__ float sh[256];
  int r = blockIdx.x, t = threadIdx.x;
  int b = r / Ln, l = r % Ln;
  const float* xr = xt + (size_t)r * Cn;
  float fs = BETAf * s_sig[(size_t)b * Ln + l] + GAMMAf * tmprl[b] + (1.0f - ALPHAf);
  float vals[3];
  float sum = 0.0f;
#pragma unroll
  for (int i = 0; i < 3; ++i) {
    int c = t + 256 * i;
    float v = xr[c] * (ALPHAf * ch[(size_t)b * Cn + c] + fs);
    vals[i] = v; sum += v;
  }
  sh[t] = sum; __syncthreads();
  for (int s = 128; s > 0; s >>= 1) { if (t < s) sh[t] += sh[t + s]; __syncthreads(); }
  float mean = sh[0] / Cn; __syncthreads();
  float sq = 0.0f;
#pragma unroll
  for (int i = 0; i < 3; ++i) { float d = vals[i] - mean; sq += d * d; }
  sh[t] = sq; __syncthreads();
  for (int s = 128; s > 0; s >>= 1) { if (t < s) sh[t] += sh[t + s]; __syncthreads(); }
  float rstd = rsqrtf(sh[0] / Cn + EPSf);
  float* op = out + (size_t)r * Cn;
#pragma unroll
  for (int i = 0; i < 3; ++i) {
    int c = t + 256 * i;
    op[c] = (vals[i] - mean) * rstd * g[c] + bb[c];
  }
}

// out[b,c,l] = gate * (LayerNorm_C(o[b,l,:]) * g + b)[c]
__global__ void k_final(const float* o, const float* g, const float* bb,
                        const float* gate, float* out) {
  __shared__ float sh[256];
  int r = blockIdx.x, t = threadIdx.x;
  int b = r / Ln, l = r % Ln;
  const float* xr = o + (size_t)r * Cn;
  float vals[3];
  float sum = 0.0f;
#pragma unroll
  for (int i = 0; i < 3; ++i) { float v = xr[t + 256 * i]; vals[i] = v; sum += v; }
  sh[t] = sum; __syncthreads();
  for (int s = 128; s > 0; s >>= 1) { if (t < s) sh[t] += sh[t + s]; __syncthreads(); }
  float mean = sh[0] / Cn; __syncthreads();
  float sq = 0.0f;
#pragma unroll
  for (int i = 0; i < 3; ++i) { float d = vals[i] - mean; sq += d * d; }
  sh[t] = sq; __syncthreads();
  for (int s = 128; s > 0; s >>= 1) { if (t < s) sh[t] += sh[t + s]; __syncthreads(); }
  float rstd = rsqrtf(sh[0] / Cn + EPSf);
  float gt = gate[0];
#pragma unroll
  for (int i = 0; i < 3; ++i) {
    int c = t + 256 * i;
    float v = ((vals[i] - mean) * rstd * g[c] + bb[c]) * gt;
    out[((size_t)b * Cn + c) * Ln + l] = v;
  }
}

// =========================================================================
extern "C" void kernel_launch(void* const* d_in, const int* in_sizes, int n_in,
                              void* d_out, int out_size, void* d_ws, size_t ws_size,
                              hipStream_t stream) {
  (void)in_sizes; (void)n_in; (void)out_size; (void)ws_size;
  const float* x       = (const float*)d_in[0];
  const float* vt      = (const float*)d_in[1];
  const float* conv_w  = (const float*)d_in[2];
  const float* conv_b  = (const float*)d_in[3];
  const float* fc_w    = (const float*)d_in[4];
  const float* fc_b    = (const float*)d_in[5];
  const float* mtk     = (const float*)d_in[6];
  const float* gate_av = (const float*)d_in[7];
  const float* a1_w    = (const float*)d_in[8];
  const float* a1_b    = (const float*)d_in[9];
  const float* v1_w    = (const float*)d_in[10];
  const float* v1_b    = (const float*)d_in[11];
  const float* bt_w    = (const float*)d_in[12];
  const float* bt_b    = (const float*)d_in[13];
  const float* vc_w    = (const float*)d_in[14];
  const float* vc_b    = (const float*)d_in[15];
  const float* v2_w    = (const float*)d_in[16];
  const float* v2_b    = (const float*)d_in[17];
  const float* a2_w    = (const float*)d_in[18];
  const float* a2_b    = (const float*)d_in[19];
  const float* vs_w    = (const float*)d_in[20];
  const float* vs_b    = (const float*)d_in[21];
  const float* tg_w    = (const float*)d_in[22];
  const float* tg_b    = (const float*)d_in[23];
  const float* lnb_g   = (const float*)d_in[24];
  const float* lnb_b   = (const float*)d_in[25];
  const float* down_w  = (const float*)d_in[26];
  const float* bn1_g   = (const float*)d_in[27];
  const float* bn1_b   = (const float*)d_in[28];
  const float* bn1_m   = (const float*)d_in[29];
  const float* bn1_v   = (const float*)d_in[30];
  const float* up_w    = (const float*)d_in[31];
  const float* bn2_g   = (const float*)d_in[32];
  const float* bn2_b   = (const float*)d_in[33];
  const float* bn2_m   = (const float*)d_in[34];
  const float* bn2_v   = (const float*)d_in[35];
  const float* lnp_g   = (const float*)d_in[36];
  const float* lnp_b   = (const float*)d_in[37];
  const float* gate    = (const float*)d_in[38];

  float* out = (float*)d_out;
  const size_t OUT_SP = (size_t)Bn * Cn * Ln;
  const size_t OUT_T  = OUT_SP + (size_t)Bn * Ln;

  // ---- workspace layout (floats) ----
  float* W = (float*)d_ws;
  size_t off = 0;
  auto alloc = [&](size_t n) { float* p = W + off; off += n; return p; };
  float* Hbuf   = alloc((size_t)Bn * NVn * NVn);      // 6.1M
  float* Vchain = alloc((size_t)Bn * NVn * Cn);       // V -> vq1 -> cfeat -> x6ln
  float* Sbuf   = alloc((size_t)Bn * Tn * NVn);       // scores1/probs1 then scores2/probs2
  float* rep    = alloc((size_t)Bn * Tn * Cn);
  float* xtbuf  = alloc((size_t)Bn * Ln * Cn);        // xt then o
  float* zbuf   = alloc((size_t)Bn * Ln * DSn);
  float* part   = alloc((size_t)Bn * Ln * NTQ);
  float* audio  = alloc((size_t)Bn * Cn);
  float* aq1    = alloc((size_t)Bn * Cn);
  float* vq1m   = alloc((size_t)Bn * Cn);
  float* prodb  = alloc((size_t)Bn * Cn);
  float* chb    = alloc((size_t)Bn * Cn);
  float* avq    = alloc((size_t)Bn * DMn);
  float* aq2    = alloc((size_t)Bn * DMn);
  float* tmprl  = alloc((size_t)Bn);
  float* s_tmp  = alloc((size_t)Bn * Ln);
  float* s_sig  = alloc((size_t)Bn * Ln);

  const int MT2 = (Ln + 31) / 32;         // 7 row blocks (32 rows each)
  const int TT2 = (Tn + 31) / 32;         // 3 row blocks for T=87
  dim3 w32(32);

  // 1) H = conv_w @ vt^T + conv_b        N=196 -> 4 strips of 64
  k_gemm_h<<<dim3(MT2, 4, Bn), w32, 0, stream>>>(conv_w, conv_b, vt, Hbuf);
  // 2) V = H @ fc_w^T + fc_b             N=768 -> 12 strips
  k_gemm_V<<<dim3(MT2, 12, Bn), w32, 0, stream>>>(Hbuf, fc_w, fc_b, Vchain);
  // 3) audio = mean_tokens(V)
  k_colmean<<<(Bn * Cn + 255) / 256, 256, 0, stream>>>(Vchain, audio, NVn, Cn, 1.0f / NVn);
  // 4) scores1 = my_tokens @ V^T ; softmax
  k_gemm_s1<<<dim3(TT2, 4, Bn), w32, 0, stream>>>(mtk, Vchain, Sbuf);
  k_softmax<<<Bn * Tn, 256, 0, stream>>>(Sbuf, NVn);
  // 5) rep = my_tokens + probs1 @ V
  k_gemm_rep<<<dim3(TT2, 12, Bn), w32, 0, stream>>>(Sbuf, Vchain, mtk, rep);
  // 6) scores2 = x^T @ rep^T ; softmax         N=87 -> 3 strips of 32
  k_gemm_s2<<<dim3(MT2, 3, Bn), w32, 0, stream>>>(x, rep, Sbuf);
  k_softmax<<<Bn * Ln, 256, 0, stream>>>(Sbuf, Tn);
  // 7) xt = x^T + gate_av * probs2 @ rep
  k_gemm_xt<<<dim3(MT2, 12, Bn), w32, 0, stream>>>(Sbuf, rep, x, gate_av, xtbuf);
  // 8) small dense layers off `audio`
  k_dense<<<(Bn * Cn + 255) / 256, 256, 0, stream>>>(audio, a1_w, a1_b, aq1, Cn, Cn, 1);
  k_dense<<<(Bn * DMn + 255) / 256, 256, 0, stream>>>(audio, a2_w, a2_b, aq2, Cn, DMn, 1);
  k_dense<<<(Bn + 255) / 256, 256, 0, stream>>>(audio, tg_w, tg_b, tmprl, Cn, 1, 2);
  // 9) vq1 = relu(xt @ v1^T + b); vq1mean = mean_L
  k_gemm_vq1<<<dim3(MT2, 12, Bn), w32, 0, stream>>>(xtbuf, v1_w, v1_b, Vchain);
  k_colmean<<<(Bn * Cn + 255) / 256, 256, 0, stream>>>(Vchain, vq1m, Ln, Cn, 1.0f / Ln);
  // 10) avq = relu((aq1*vq1mean) @ bt^T); ch = sigmoid(avq @ vc^T)
  k_prod<<<(Bn * Cn + 255) / 256, 256, 0, stream>>>(aq1, vq1m, prodb, Bn * Cn);
  k_dense<<<(Bn * DMn + 255) / 256, 256, 0, stream>>>(prodb, bt_w, bt_b, avq, Cn, DMn, 1);
  k_dense<<<(Bn * Cn + 255) / 256, 256, 0, stream>>>(avq, vc_w, vc_b, chb, DMn, Cn, 2);
  // 11) cfeat = xt*(ch+1); cq GEMM fused with (cq*aq2)@vs_w row-reduction
  k_cfeat<<<(Bn * Ln * Cn + 255) / 256, 256, 0, stream>>>(xtbuf, chb, Vchain);
  k_gemm_cq<<<dim3(MT2, 6, Bn), w32, 0, stream>>>(Vchain, v2_w, v2_b, aq2, vs_w, part);
  k_stmp<<<(Bn * Ln + 255) / 256, 256, 0, stream>>>(part, vs_b, s_tmp);
  // 12) spatial softmax (output 2) + s_sig; temporal (output 3)
  k_spatial<<<Bn, 256, 0, stream>>>(s_tmp, s_sig, out + OUT_SP);
  k_copy<<<(Bn + 255) / 256, 256, 0, stream>>>(tmprl, out + OUT_T, Bn);
  // 13) x6 = xt*factor, LN -> x6ln
  k_x6ln<<<Bn * Ln, 256, 0, stream>>>(xtbuf, chb, s_sig, tmprl, lnb_g, lnb_b, Vchain);
  // 14) bottleneck adapter
  k_gemm_down<<<dim3(MT2, 1, Bn), w32, 0, stream>>>(Vchain, down_w, bn1_g, bn1_b,
                                                    bn1_m, bn1_v, zbuf);
  k_gemm_up<<<dim3(MT2, 12, Bn), w32, 0, stream>>>(zbuf, up_w, bn2_g, bn2_b,
                                                   bn2_m, bn2_v, xtbuf);
  // 15) final LN + gate + transpose (output 1)
  k_final<<<Bn * Ln, 256, 0, stream>>>(xtbuf, lnp_g, lnp_b, gate, out);
}
